// MultibandFrameAttention_14362370638501
// MI455X (gfx1250) — matrix operand order repaired and checked by reading, not compiled, not run on hardware
//
#include <hip/hip_runtime.h>
#include <hip/hip_bf16.h>

// ---------------------------------------------------------------------------
// MultibandFrameAttention for MI455X (gfx1250, wave32, WMMA)
//
// a[r,c] = (q_r . k_c  +  er[:,r] . q_c) / 32          (skew trick resolved)
// o      = softmax(a) @ V @ Wo^T + bo
//
// All big matmuls use f32-emulated bf16 WMMA: x = hi + lo (bf16 each),
// C += Ah*Bh + Ah*Bl + Al*Bh  via v_wmma_f32_16x16x32_bf16 (3 per K=32).
// ---------------------------------------------------------------------------

typedef __bf16 bf16_t;
typedef bf16_t bf16x8  __attribute__((ext_vector_type(8)));
typedef bf16_t bf16x16 __attribute__((ext_vector_type(16)));
typedef float  f32x8   __attribute__((ext_vector_type(8)));
typedef float  v8f     __attribute__((ext_vector_type(8)));

#define NB 2
#define WW 2048
#define BINS_ 1024
#define BANDS_ 8
#define HEAD_ 128

// ---- WMMA operand loaders (ISA 7.12.2 layouts, wave32) --------------------
// A (16x32 bf16, M x K): lane m=l&15 row M=m; g=l>>4 selects K chunks
//   elements 0..7  -> K = k0 + g*8 .. +7
//   elements 8..15 -> K = k0 + 16 + g*8 .. +7
__device__ inline bf16x16 load_a(const bf16_t* __restrict__ base, int ldk,
                                 int row0, int k0, int lane) {
  int g = lane >> 4, m = lane & 15;
  const bf16_t* p = base + (size_t)(row0 + m) * ldk + k0 + g * 8;
  bf16x8 c0 = *(const bf16x8*)p;
  bf16x8 c1 = *(const bf16x8*)(p + 16);
  bf16x16 r;
#pragma unroll
  for (int i = 0; i < 8; ++i) { r[i] = c0[i]; r[i + 8] = c1[i]; }
  return r;
}

// B (32x16 bf16, K x N) loaded from row-major [N, K] buffer:
// lane n=l&15 column N=n; elements 0..7 -> K = k0 + g*8 .. +7,
//   elements 8..15 -> K = k0 + 16 + g*8 .. +7  (g = l>>4)
__device__ inline bf16x16 load_b(const bf16_t* __restrict__ base, int ldk,
                                 int col0, int k0, int lane) {
  int g = lane >> 4, n = lane & 15;
  const bf16_t* p = base + (size_t)(col0 + n) * ldk + k0 + g * 8;
  bf16x8 c0 = *(const bf16x8*)p;
  bf16x8 c1 = *(const bf16x8*)(p + 16);
  bf16x16 r;
#pragma unroll
  for (int i = 0; i < 8; ++i) { r[i] = c0[i]; r[i + 8] = c1[i]; }
  return r;
}

__device__ inline v8f wmma3(v8f c, bf16x16 ah, bf16x16 al,
                            bf16x16 bh, bf16x16 bl) {
  c = __builtin_amdgcn_wmma_f32_16x16x32_bf16(false, ah, false, bh, (short)0, c, false, false);
  c = __builtin_amdgcn_wmma_f32_16x16x32_bf16(false, ah, false, bl, (short)0, c, false, false);
  c = __builtin_amdgcn_wmma_f32_16x16x32_bf16(false, al, false, bh, (short)0, c, false, false);
  return c;
}

__device__ inline void split2(float x, bf16_t& h, bf16_t& l) {
  h = (bf16_t)x;
  l = (bf16_t)(x - (float)h);
}

// ---- elementwise f32 -> (hi,lo) bf16 split --------------------------------
__global__ void split_kernel(const float* __restrict__ src,
                             bf16_t* __restrict__ h, bf16_t* __restrict__ l,
                             int n) {
  int i = blockIdx.x * blockDim.x + threadIdx.x;
  if (i < n) {
    bf16_t hh, ll;
    split2(src[i], hh, ll);
    h[i] = hh; l[i] = ll;
  }
}

// er [HEAD, W] -> ErT [W, HEAD] hi/lo  (row r of ErT = er[:, r])
__global__ void er_transpose_kernel(const float* __restrict__ er,
                                    bf16_t* __restrict__ h, bf16_t* __restrict__ l) {
  int i = blockIdx.x * blockDim.x + threadIdx.x;
  if (i < WW * HEAD_) {
    int c = i >> 7, d = i & 127;
    bf16_t hh, ll;
    split2(er[(size_t)d * WW + c], hh, ll);
    h[i] = hh; l[i] = ll;
  }
}

// ---- generic 4096x1024x1024 GEMM  out = A @ B^T + bias --------------------
// mode 0: store q  [b,n,w,d] hi/lo      mode 1: store k  [b,n,w,d] hi/lo
// mode 2: store vT [b,n,d,w] hi/lo      mode 3: store f32 to outF [m, 1024]
__global__ __launch_bounds__(256) void gemm_split_kernel(
    const bf16_t* __restrict__ Ah, const bf16_t* __restrict__ Al,
    const bf16_t* __restrict__ Bh, const bf16_t* __restrict__ Bl,
    const float* __restrict__ bias,
    bf16_t* __restrict__ outH, bf16_t* __restrict__ outL,
    float* __restrict__ outF, int mode) {
  const int Kd = 1024;
  int lane = threadIdx.x & 31, wave = threadIdx.x >> 5;
  int tile = blockIdx.x * 8 + wave;          // 256*64 tiles
  int tN = tile & 63, tM = tile >> 6;
  int row0 = tM * 16, col0 = tN * 16;
  v8f c = {};
  for (int k0 = 0; k0 < Kd; k0 += 32) {
    bf16x16 ah = load_a(Ah, Kd, row0, k0, lane);
    bf16x16 al = load_a(Al, Kd, row0, k0, lane);
    bf16x16 bh = load_b(Bh, Kd, col0, k0, lane);
    bf16x16 bl = load_b(Bl, Kd, col0, k0, lane);
    c = wmma3(c, ah, al, bh, bl);
  }
  int g = lane >> 4, nn = lane & 15;
  int o = col0 + nn;
  float bval = bias[o];
#pragma unroll
  for (int v = 0; v < 8; ++v) {
    int m = row0 + g * 8 + v;
    float val = c[v] + bval;
    if (mode == 3) {
      outF[(size_t)m * 1024 + o] = val;
    } else {
      int b = m >> 11, w = m & 2047;
      int nb = o >> 7, d = o & 127;
      size_t idx = (mode == 2)
          ? (((size_t)(b * BANDS_ + nb) * HEAD_ + d) * WW + w)
          : (((size_t)(b * BANDS_ + nb) * WW + w) * HEAD_ + d);
      bf16_t h, l; split2(val, h, l);
      outH[idx] = h; outL[idx] = l;
    }
  }
}

// ---- scores: a = (Q K^T + ErT Q^T) / 32 -----------------------------------
__global__ __launch_bounds__(256) void scores_kernel(
    const bf16_t* __restrict__ qh, const bf16_t* __restrict__ ql,
    const bf16_t* __restrict__ kh, const bf16_t* __restrict__ kl,
    const bf16_t* __restrict__ erh, const bf16_t* __restrict__ erl,
    float* __restrict__ aOut) {
  int lane = threadIdx.x & 31, wave = threadIdx.x >> 5;
  int tile = blockIdx.x * 8 + wave;          // 16 * 128 * 128 tiles
  int ct = tile & 127;
  int rt = (tile >> 7) & 127;
  int bn = tile >> 14;
  const bf16_t* qbh = qh + (size_t)bn * WW * HEAD_;
  const bf16_t* qbl = ql + (size_t)bn * WW * HEAD_;
  const bf16_t* kbh = kh + (size_t)bn * WW * HEAD_;
  const bf16_t* kbl = kl + (size_t)bn * WW * HEAD_;
  v8f c = {};
  for (int k0 = 0; k0 < HEAD_; k0 += 32) {
    // term 1: Q rows x K cols
    bf16x16 a1h = load_a(qbh, HEAD_, rt * 16, k0, lane);
    bf16x16 a1l = load_a(qbl, HEAD_, rt * 16, k0, lane);
    bf16x16 b1h = load_b(kbh, HEAD_, ct * 16, k0, lane);
    bf16x16 b1l = load_b(kbl, HEAD_, ct * 16, k0, lane);
    c = wmma3(c, a1h, a1l, b1h, b1l);
    // term 2 (skew bias): ErT rows x Q cols
    bf16x16 a2h = load_a(erh, HEAD_, rt * 16, k0, lane);
    bf16x16 a2l = load_a(erl, HEAD_, rt * 16, k0, lane);
    bf16x16 b2h = load_b(qbh, HEAD_, ct * 16, k0, lane);
    bf16x16 b2l = load_b(qbl, HEAD_, ct * 16, k0, lane);
    c = wmma3(c, a2h, a2l, b2h, b2l);
  }
  int g = lane >> 4, nn = lane & 15;
  size_t base = (size_t)bn * WW * WW;
#pragma unroll
  for (int v = 0; v < 8; ++v) {
    int r = rt * 16 + g * 8 + v;
    int col = ct * 16 + nn;
    aOut[base + (size_t)r * WW + col] = c[v] * 0.03125f;  // 1/sqrt(1024)
  }
}

// ---- online-softmax row stats: (max, 1/sum) per row -----------------------
__global__ __launch_bounds__(256) void softmax_stats_kernel(
    const float* __restrict__ aOut, float* __restrict__ stats) {
  int lane = threadIdx.x & 31, wave = threadIdx.x >> 5;
  int row = blockIdx.x * 8 + wave;           // 32768 rows
  const float* p = aOut + (size_t)row * WW;
  float m = -3.4e38f, s = 0.f;
  for (int i = lane; i < WW; i += 32) {
    float x = p[i];
    float mn = fmaxf(m, x);
    s = s * __expf(m - mn) + __expf(x - mn);
    m = mn;
  }
  for (int off = 16; off; off >>= 1) {
    float m2 = __shfl_xor(m, off, 32);
    float s2 = __shfl_xor(s, off, 32);
    float mn = fmaxf(m, m2);
    s = s * __expf(m - mn) + s2 * __expf(m2 - mn);
    m = mn;
  }
  if (lane == 0) { stats[2 * row] = m; stats[2 * row + 1] = 1.0f / s; }
}

// ---- ctx = softmax(a) @ V, attn recomputed on the fly ---------------------
__global__ __launch_bounds__(256) void ctx_kernel(
    const float* __restrict__ aOut, const float* __restrict__ stats,
    const bf16_t* __restrict__ vth, const bf16_t* __restrict__ vtl,
    bf16_t* __restrict__ ctxh, bf16_t* __restrict__ ctxl) {
  int lane = threadIdx.x & 31, wave = threadIdx.x >> 5;
  int tile = blockIdx.x * 8 + wave;          // 16 * 128 * 8 tiles
  int dt = tile & 7;
  int rt = (tile >> 3) & 127;
  int bn = tile >> 10;
  int g = lane >> 4, idx15 = lane & 15;
  int r = rt * 16 + idx15;                   // this lane's A row
  size_t rowbase = ((size_t)bn * WW + r) * WW;
  float mx  = stats[2 * (bn * WW + r)];
  float inv = stats[2 * (bn * WW + r) + 1];
  const bf16_t* vbh = vth + (size_t)bn * HEAD_ * WW;
  const bf16_t* vbl = vtl + (size_t)bn * HEAD_ * WW;
  v8f c = {};
  for (int k0 = 0; k0 < WW; k0 += 32) {
    f32x8 x0 = *(const f32x8*)(aOut + rowbase + k0 + g * 8);
    f32x8 x1 = *(const f32x8*)(aOut + rowbase + k0 + 16 + g * 8);
    bf16x16 ah, al;
#pragma unroll
    for (int i = 0; i < 8; ++i) {
      float e0 = __expf(x0[i] - mx) * inv;
      bf16_t h0 = (bf16_t)e0; ah[i] = h0; al[i] = (bf16_t)(e0 - (float)h0);
      float e1 = __expf(x1[i] - mx) * inv;
      bf16_t h1 = (bf16_t)e1; ah[i + 8] = h1; al[i + 8] = (bf16_t)(e1 - (float)h1);
    }
    bf16x16 bh = load_b(vbh, WW, dt * 16, k0, lane);
    bf16x16 bl = load_b(vbl, WW, dt * 16, k0, lane);
    c = wmma3(c, ah, al, bh, bl);
  }
  int b = bn >> 3, nb = bn & 7;
#pragma unroll
  for (int v = 0; v < 8; ++v) {
    int rr = rt * 16 + g * 8 + v;
    int d = dt * 16 + idx15;
    size_t o = ((size_t)(b * WW + rr)) * BINS_ + nb * HEAD_ + d;
    bf16_t h, l; split2(c[v], h, l);
    ctxh[o] = h; ctxl[o] = l;
  }
}

// ---------------------------------------------------------------------------
extern "C" void kernel_launch(void* const* d_in, const int* in_sizes, int n_in,
                              void* d_out, int out_size, void* d_ws, size_t ws_size,
                              hipStream_t stream) {
  const float* x  = (const float*)d_in[0];
  const float* Wq = (const float*)d_in[1];
  const float* bq = (const float*)d_in[2];
  const float* Wk = (const float*)d_in[3];
  const float* bk = (const float*)d_in[4];
  const float* Wv = (const float*)d_in[5];
  const float* bv = (const float*)d_in[6];
  const float* Wo = (const float*)d_in[7];
  const float* bo = (const float*)d_in[8];
  const float* er = (const float*)d_in[9];

  float* oOut = (float*)d_out;                         // [2,2048,1024]
  float* aOut = (float*)d_out + (size_t)NB * WW * BINS_; // [2,8,2048,2048]

  // workspace carve-out
  char* p = (char*)d_ws;
  auto alloc = [&](size_t bytes) -> void* {
    void* r = (void*)p;
    p += (bytes + 255) & ~(size_t)255;
    return r;
  };
  const size_t XN  = (size_t)NB * WW * BINS_;    // 4096*1024
  const size_t WN  = (size_t)BINS_ * BINS_;      // 1024*1024
  const size_t QN  = (size_t)NB * BANDS_ * WW * HEAD_;
  bf16_t* xh  = (bf16_t*)alloc(XN * 2);  bf16_t* xl  = (bf16_t*)alloc(XN * 2);
  bf16_t* wqh = (bf16_t*)alloc(WN * 2);  bf16_t* wql = (bf16_t*)alloc(WN * 2);
  bf16_t* wkh = (bf16_t*)alloc(WN * 2);  bf16_t* wkl = (bf16_t*)alloc(WN * 2);
  bf16_t* wvh = (bf16_t*)alloc(WN * 2);  bf16_t* wvl = (bf16_t*)alloc(WN * 2);
  bf16_t* woh = (bf16_t*)alloc(WN * 2);  bf16_t* wol = (bf16_t*)alloc(WN * 2);
  bf16_t* erh = (bf16_t*)alloc((size_t)WW * HEAD_ * 2);
  bf16_t* erl = (bf16_t*)alloc((size_t)WW * HEAD_ * 2);
  bf16_t* qh  = (bf16_t*)alloc(QN * 2);  bf16_t* ql  = (bf16_t*)alloc(QN * 2);
  bf16_t* kh  = (bf16_t*)alloc(QN * 2);  bf16_t* kl  = (bf16_t*)alloc(QN * 2);
  bf16_t* vth = (bf16_t*)alloc(QN * 2);  bf16_t* vtl = (bf16_t*)alloc(QN * 2);
  bf16_t* ctxh = (bf16_t*)alloc(XN * 2); bf16_t* ctxl = (bf16_t*)alloc(XN * 2);
  float*  stats = (float*)alloc((size_t)NB * BANDS_ * WW * 2 * 4);

  // 1) hi/lo splits
  split_kernel<<<(int)((XN + 255) / 256), 256, 0, stream>>>(x, xh, xl, (int)XN);
  split_kernel<<<(int)((WN + 255) / 256), 256, 0, stream>>>(Wq, wqh, wql, (int)WN);
  split_kernel<<<(int)((WN + 255) / 256), 256, 0, stream>>>(Wk, wkh, wkl, (int)WN);
  split_kernel<<<(int)((WN + 255) / 256), 256, 0, stream>>>(Wv, wvh, wvl, (int)WN);
  split_kernel<<<(int)((WN + 255) / 256), 256, 0, stream>>>(Wo, woh, wol, (int)WN);
  er_transpose_kernel<<<(WW * HEAD_ + 255) / 256, 256, 0, stream>>>(er, erh, erl);

  // 2) projections: 4096x1024x1024 each, wave-per-16x16-tile, 8 waves/block
  gemm_split_kernel<<<2048, 256, 0, stream>>>(xh, xl, wqh, wql, bq, qh,  ql,  nullptr, 0);
  gemm_split_kernel<<<2048, 256, 0, stream>>>(xh, xl, wkh, wkl, bk, kh,  kl,  nullptr, 1);
  gemm_split_kernel<<<2048, 256, 0, stream>>>(xh, xl, wvh, wvl, bv, vth, vtl, nullptr, 2);

  // 3) scores (writes a directly to d_out)
  scores_kernel<<<32768, 256, 0, stream>>>(qh, ql, kh, kl, erh, erl, aOut);

  // 4) softmax row stats
  softmax_stats_kernel<<<4096, 256, 0, stream>>>(aOut, stats);

  // 5) ctx = softmax(a) @ V
  ctx_kernel<<<2048, 256, 0, stream>>>(aOut, stats, vth, vtl, ctxh, ctxl);

  // 6) o = ctx @ Wo^T + bo
  gemm_split_kernel<<<2048, 256, 0, stream>>>(ctxh, ctxl, woh, wol, bo,
                                              nullptr, nullptr, oOut, 3);
}
